// ScorePredictor_3564822856254
// MI455X (gfx1250) — matrix operand order repaired and checked by reading, not compiled, not run on hardware
//
#include <hip/hip_runtime.h>
#include <math.h>

// CDNA5 / gfx1250: wave32, WMMA 16x16x32 bf16 with f32 accumulate.
typedef __attribute__((ext_vector_type(16))) __bf16        v16bf;
typedef __attribute__((ext_vector_type(2)))  __bf16        v2bf;
typedef __attribute__((ext_vector_type(8)))  float         v8f;

#define IN_F   64
#define PITCH1 200   // halfs per W1 row in LDS (400 B, 16B-aligned, bank-spread)
#define PITCH2 72    // halfs per W2 / h1 row in LDS (144 B, 16B-aligned, bank-spread)

// ---- fast f32 -> bf16: hardware packed convert (confirmed: v_cvt_pk_bf16_f32) ----
#if __has_builtin(__builtin_amdgcn_cvt_pk_bf16_f32)
__device__ __forceinline__ v2bf pk2(float lo, float hi) {
  return __builtin_amdgcn_cvt_pk_bf16_f32(lo, hi);
}
#else
__device__ __forceinline__ v2bf pk2(float lo, float hi) {
  v2bf r; r[0] = (__bf16)lo; r[1] = (__bf16)hi; return r;
}
#endif
__device__ __forceinline__ unsigned short bf16_1(float f) {
  return __builtin_bit_cast(unsigned short, (__bf16)f);
}

// single-instruction ReLU: v_med3_f32(x, 0, +inf)
#if __has_builtin(__builtin_amdgcn_fmed3f)
__device__ __forceinline__ float relu1(float x) {
  return __builtin_amdgcn_fmed3f(x, 0.0f, __builtin_inff());
}
#else
__device__ __forceinline__ float relu1(float x) { return fmaxf(x, 0.0f); }
#endif

union frag16 { v2bf h[8]; v16bf v; };

// load two 8-half bf16 runs [p,p+8) and [p+16,p+24) from LDS as a WMMA fragment
__device__ __forceinline__ v16bf ld_frag(const unsigned short* p) {
  struct Raw { uint4 a, b; } r;
  r.a = *(const uint4*)p;
  r.b = *(const uint4*)(p + 16);
  return __builtin_bit_cast(v16bf, r);
}

__global__ __launch_bounds__(128) void edge_mlp_kernel(
    const float* __restrict__ x,  const float* __restrict__ e,
    const int*   __restrict__ src, const int* __restrict__ dst,
    const float* __restrict__ W1, const float* __restrict__ b1,
    const float* __restrict__ W2, const float* __restrict__ b2,
    const float* __restrict__ W3, const float* __restrict__ b3,
    float* __restrict__ out, int nTiles, int nEdges)
{
  __shared__ unsigned short sW1[64 * PITCH1];     // 25.0 KB  W1 bf16 row-major
  __shared__ unsigned short sW2[64 * PITCH2];     //  9.0 KB  W2 bf16 row-major
  __shared__ unsigned short sH1[4][16 * PITCH2];  //  9.0 KB  per-wave h1 scratch

  const int tid = threadIdx.x;
  // ---- stage weights into LDS as bf16 (once per block, amortized) ----
  for (int i = tid; i < 64 * 192; i += blockDim.x)
    sW1[(i / 192) * PITCH1 + (i % 192)] = bf16_1(W1[i]);
  for (int i = tid; i < 64 * 64; i += blockDim.x)
    sW2[(i / 64) * PITCH2 + (i % 64)] = bf16_1(W2[i]);
  __syncthreads();

  const int lane = tid & 31;
  const int m    = lane & 15;   // A: row-in-tile; B: column-in-n-tile
  const int half = lane >> 4;   // K-half selector of the WMMA layouts
  const int waveInBlk = tid >> 5;
  unsigned short* sH = sH1[waveInBlk];

  // per-lane resident biases / w3 slice (N = nt*16 + m)
  float b1v[4], b2v[4], w3v[4];
  #pragma unroll
  for (int nt = 0; nt < 4; ++nt) {
    b1v[nt] = b1[nt * 16 + m];
    b2v[nt] = b2[nt * 16 + m];
    w3v[nt] = W3[nt * 16 + m];
  }
  const float bias3 = b3[0];

  const int wave   = blockIdx.x * (blockDim.x >> 5) + waveInBlk;
  const int nWaves = gridDim.x * (blockDim.x >> 5);

  // ---- software pipeline: indices for the first tile ----
  int tile = wave;
  int sIdx = 0, dIdx = 0;
  if (tile < nTiles) {
    int ec = tile * 16 + m; if (ec >= nEdges) ec = nEdges - 1;
    sIdx = src[ec];
    dIdx = dst[ec];
  }

  for (; tile < nTiles; tile += nWaves) {
    int edge = tile * 16 + m;
    int eclamp = edge < nEdges ? edge : nEdges - 1;  // tail-safe (EXEC stays full for WMMA)
    const float* seg[3];
    seg[0] = x + (size_t)sIdx * IN_F;
    seg[1] = x + (size_t)dIdx * IN_F;
    seg[2] = e + (size_t)eclamp * IN_F;

    // kick off NEXT tile's index loads now; they complete under this tile's math
    const int nTile = tile + nWaves;                  // wave-uniform
    int nClamp = nTile * 16 + m; if (nClamp >= nEdges) nClamp = nEdges - 1;
    int nsIdx = sIdx, ndIdx = dIdx;
    if (nTile < nTiles) {                             // uniform branch
      nsIdx = src[nClamp];
      ndIdx = dst[nClamp];
    }

    // ---- build layer-1 A fragments: 16x192 bf16 = 6 chunks of 16x32 ----
    // lane holds K runs [base,base+8) and [base+16,base+24), base = kc*32 + half*8
    v16bf A1[6];
    #pragma unroll
    for (int kc = 0; kc < 6; ++kc) {
      const float* sp = seg[kc >> 1];            // chunk fits inside one 64-f32 segment
      const int off = (kc * 32 + half * 8) & 63;
      float4 r0a = *(const float4*)(sp + off);
      float4 r0b = *(const float4*)(sp + off + 4);
      float4 r1a = *(const float4*)(sp + off + 16);
      float4 r1b = *(const float4*)(sp + off + 20);
      frag16 u;
      u.h[0] = pk2(r0a.x, r0a.y); u.h[1] = pk2(r0a.z, r0a.w);
      u.h[2] = pk2(r0b.x, r0b.y); u.h[3] = pk2(r0b.z, r0b.w);
      u.h[4] = pk2(r1a.x, r1a.y); u.h[5] = pk2(r1a.z, r1a.w);
      u.h[6] = pk2(r1b.x, r1b.y); u.h[7] = pk2(r1b.z, r1b.w);
      A1[kc] = u.v;
    }

    // ---- layer 1: 4 n-tiles x 6 k-chunks of v_wmma_f32_16x16x32_bf16 ----
    #pragma unroll
    for (int nt = 0; nt < 4; ++nt) {
      const float bv = b1v[nt];                  // bias folded into C init (all 8 VGPRs share N)
      v8f c = {bv, bv, bv, bv, bv, bv, bv, bv};
      const unsigned short* wrow = &sW1[(nt * 16 + m) * PITCH1 + half * 8];
      // double-buffer B fragments so each WMMA overlaps the next ds_load pair
      v16bf bcur = ld_frag(wrow);
      #pragma unroll
      for (int kc = 0; kc < 6; ++kc) {
        v16bf bnxt = bcur;
        if (kc < 5) bnxt = ld_frag(wrow + (kc + 1) * 32);
        c = __builtin_amdgcn_wmma_f32_16x16x32_bf16(
                false, A1[kc], false, bcur, (short)0, c, false, false);
        bcur = bnxt;
      }
      // ReLU + bf16 -> h1 scratch (D layout: VGPR r -> M = r + 8*half, N = nt*16+m)
      #pragma unroll
      for (int r = 0; r < 8; ++r)
        sH[(r + 8 * half) * PITCH2 + nt * 16 + m] = bf16_1(relu1(c[r]));
    }

    // prefetch next tile's feature rows (256 B each) into cache under the WMMAs
    if (nTile < nTiles) {                             // uniform branch
      const float* p0 = x + (size_t)nsIdx * IN_F;
      const float* p1 = x + (size_t)ndIdx * IN_F;
      const float* p2 = e + (size_t)nClamp * IN_F;
      __builtin_prefetch(p0, 0, 3); __builtin_prefetch(p0 + 32, 0, 3);
      __builtin_prefetch(p1, 0, 3); __builtin_prefetch(p1 + 32, 0, 3);
      __builtin_prefetch(p2, 0, 3); __builtin_prefetch(p2 + 32, 0, 3);
    }

    asm volatile("s_wait_dscnt 0" ::: "memory");   // h1 stores visible before re-read

    // ---- layer-2 A fragments from h1 scratch (16x64 bf16 = 2 chunks) ----
    v16bf A2[2];
    #pragma unroll
    for (int kc = 0; kc < 2; ++kc) {
      const unsigned short* hp = &sH[m * PITCH2 + kc * 32 + half * 8];
      struct Raw { uint4 a, b; } r;
      r.a = *(const uint4*)(hp);
      r.b = *(const uint4*)(hp + 16);              // second K run is +16 halfs away
      A2[kc] = __builtin_bit_cast(v16bf, r);
    }

    float acc[8];
    #pragma unroll
    for (int r = 0; r < 8; ++r) acc[r] = 0.f;

    // ---- layer 2 WMMA + fused layer-3 partial dot (N_CLS = 1) ----
    #pragma unroll
    for (int nt = 0; nt < 4; ++nt) {
      const float bv = b2v[nt];
      v8f c = {bv, bv, bv, bv, bv, bv, bv, bv};
      const unsigned short* wrow = &sW2[(nt * 16 + m) * PITCH2 + half * 8];
      v16bf bk0 = ld_frag(wrow);            // both fragments in flight before first WMMA
      v16bf bk1 = ld_frag(wrow + 32);
      c = __builtin_amdgcn_wmma_f32_16x16x32_bf16(
              false, A2[0], false, bk0, (short)0, c, false, false);
      c = __builtin_amdgcn_wmma_f32_16x16x32_bf16(
              false, A2[1], false, bk1, (short)0, c, false, false);
      const float w3s = w3v[nt];
      #pragma unroll
      for (int r = 0; r < 8; ++r)
        acc[r] += relu1(c[r]) * w3s;              // this lane's N column contribution
    }

    // reduce over the 16 N columns spread across lanes of each half-group
    #pragma unroll
    for (int r = 0; r < 8; ++r) {
      float v = acc[r];
      v += __shfl_xor(v, 1, 32);
      v += __shfl_xor(v, 2, 32);
      v += __shfl_xor(v, 4, 32);
      v += __shfl_xor(v, 8, 32);
      acc[r] = v;
    }
    if (m == 0) {                                 // lanes 0 (M=0..7) and 16 (M=8..15)
      #pragma unroll
      for (int r = 0; r < 8; ++r) {
        int oi = tile * 16 + 8 * half + r;
        if (oi < nEdges) out[oi] = acc[r] + bias3;
      }
    }

    sIdx = nsIdx;   // rotate the pipeline
    dIdx = ndIdx;
  }
}

extern "C" void kernel_launch(void* const* d_in, const int* in_sizes, int n_in,
                              void* d_out, int out_size, void* d_ws, size_t ws_size,
                              hipStream_t stream) {
  const float* x  = (const float*)d_in[0];
  const float* e  = (const float*)d_in[1];
  const int*   src = (const int*)d_in[2];
  const int*   dst = (const int*)d_in[3];
  const float* W1 = (const float*)d_in[4];
  const float* b1 = (const float*)d_in[5];
  const float* W2 = (const float*)d_in[6];
  const float* b2 = (const float*)d_in[7];
  const float* W3 = (const float*)d_in[8];
  const float* b3 = (const float*)d_in[9];
  float* out = (float*)d_out;

  const int nEdges = in_sizes[2];              // length of src
  const int nTiles = (nEdges + 15) / 16;       // 78125 for the reference sizes
  int blocks = 2048;                           // persistent grid-stride waves
  int maxBlocks = (nTiles + 3) / 4;            // 4 waves per 128-thread block
  if (blocks > maxBlocks) blocks = maxBlocks;
  if (blocks < 1) blocks = 1;

  edge_mlp_kernel<<<blocks, 128, 0, stream>>>(x, e, src, dst,
                                              W1, b1, W2, b2, W3, b3,
                                              out, nTiles, nEdges);
}
